// RefineOcc_21560735826370
// MI455X (gfx1250) — matrix operand order repaired and checked by reading, not compiled, not run on hardware
//
#include <hip/hip_runtime.h>

typedef __attribute__((ext_vector_type(16))) _Float16 v16h;
typedef __attribute__((ext_vector_type(8)))  _Float16 v8h;
typedef __attribute__((ext_vector_type(8)))  float    v8f;

#define B_   4
#define H_   256
#define W_   512
#define HP_  258
#define WP_  514
#define PP_  (B_*HP_*WP_)      /* 530448 padded pixels */
#define CSTR 128               /* channel stride (halves) in activation buffers */
#define NPIX (B_*H_*W_)        /* 524288 interior pixels */

#if defined(__has_builtin)
#if __has_builtin(__builtin_amdgcn_sched_group_barrier)
#define SGB(mask, cnt) __builtin_amdgcn_sched_group_barrier((mask), (cnt), 0)
#endif
#endif
#ifndef SGB
#define SGB(mask, cnt)
#endif

// ---------------------------------------------------------------------------
// Input packing: fp32 NCHW (occ[1] | feat1[32] | feat2[32]) -> f16 [pixel][C]
// with zero spatial halo. Also zeroes the halo of the second ping-pong buffer.
// ---------------------------------------------------------------------------
__global__ __launch_bounds__(256)
void init_pack_kernel(const float* __restrict__ occ,
                      const float* __restrict__ f1,
                      const float* __restrict__ f2,
                      _Float16* __restrict__ bufA,
                      _Float16* __restrict__ bufB)
{
    int p = blockIdx.x * 256 + threadIdx.x;
    if (p >= PP_) return;
    int b   = p / (HP_ * WP_);
    int rem = p - b * (HP_ * WP_);
    int yy  = rem / WP_;
    int xx  = rem - yy * WP_;
    _Float16* a = bufA + (size_t)p * CSTR;
    if (yy == 0 || yy == HP_ - 1 || xx == 0 || xx == WP_ - 1) {
        _Float16* bb = bufB + (size_t)p * CSTR;
        v8h z = {};
#pragma unroll
        for (int i = 0; i < 16; ++i) { ((v8h*)a)[i] = z; ((v8h*)bb)[i] = z; }
    } else {
        int y = yy - 1, x = xx - 1;
        const size_t HW = (size_t)H_ * W_;
        size_t qi = (size_t)y * W_ + x;
        _Float16 tmp[96];
        tmp[0] = (_Float16)occ[(size_t)b * HW + qi];
#pragma unroll
        for (int c = 0; c < 32; ++c)
            tmp[1 + c]  = (_Float16)f1[((size_t)(b * 32 + c)) * HW + qi];
#pragma unroll
        for (int c = 0; c < 32; ++c)
            tmp[33 + c] = (_Float16)f2[((size_t)(b * 32 + c)) * HW + qi];
#pragma unroll
        for (int c = 65; c < 96; ++c) tmp[c] = (_Float16)0.f;
#pragma unroll
        for (int i = 0; i < 12; ++i) ((v8h*)a)[i] = *(const v8h*)(tmp + i * 8);
    }
}

// ---------------------------------------------------------------------------
// Weight packing: fp32 OIHW -> f16 fragments laid out in per-lane A-matrix
// (16x32 f16) order with contiguous k per lane half (as B uses), so the conv kernel loads one v16h per lane.
// Fragment order: ((tap*NCH + chunk)*NMT + mt), within: lane*16 + t where
//   K(t,lane) = t + ((lane>>4)<<4),  co = mt*16 + (lane&15)
// ---------------------------------------------------------------------------
__global__ __launch_bounds__(256)
void pack_w_kernel(const float* __restrict__ w, _Float16* __restrict__ dst,
                   int Cin, int Cout, int nch, int nmt, int total)
{
    int idx = blockIdx.x * 256 + threadIdx.x;
    if (idx >= total) return;
    int t    = idx & 15;
    int r    = idx >> 4;
    int lane = r & 31;  r >>= 5;
    int mt   = r % nmt; r /= nmt;
    int chnk = r % nch;
    int tap  = r / nch;
    int K  = t | ((lane >> 4) << 4);
    int ci = chnk * 32 + K;
    int co = mt * 16 + (lane & 15);
    float v = (ci < Cin && co < Cout) ? w[((size_t)co * Cin + ci) * 9 + tap] : 0.f;
    dst[idx] = (_Float16)v;
}

// ---------------------------------------------------------------------------
// Fused implicit-GEMM conv + bias + LeakyReLU(0.1), f16 in/out, f32 accum.
// Wave = 16 pixels (N) x NMT*16 output channels (M); K = 9 taps x NCH*32 ch.
// Weights staged per-tap into double-buffered LDS (<=64KB), prefetched one
// tap ahead. A-fragments stream from LDS; sched_group_barrier pins the
// machine order to [wmma_k, ds_load(k+1) x2] so the next fragment's DS reads
// are in flight while the current WMMA executes in the XDL pipe.
// __launch_bounds__(256, 4): occupancy is LDS-limited (~10 waves/SIMD), so
// relax the VGPR ceiling to let both fragment buffers + bf[] stay live.
// ---------------------------------------------------------------------------
template<int NCH, int NMT>
__global__ __launch_bounds__(256, 4)
void conv_kernel(const _Float16* __restrict__ in, _Float16* __restrict__ out,
                 const _Float16* __restrict__ wp, const float* __restrict__ bias,
                 int Cout)
{
    constexpr int TAPH  = NCH * NMT * 512;   // halves per tap
    constexpr int TAPV8 = TAPH / 8;          // 16B units per tap
    constexpr int KFRAG = NCH * NMT;         // fragments per tap
    __shared__ __align__(32) _Float16 lds[2][TAPH];

    const int tid  = threadIdx.x;
    const int lane = tid & 31;
    const int wave = tid >> 5;
    const int q = (blockIdx.x * 8 + wave) * 16 + (lane & 15);
    const int x  = q & (W_ - 1);
    const int yb = q >> 9;
    const int y  = yb & (H_ - 1);
    const int b  = yb >> 8;
    const size_t p = (size_t)(b * HP_ + y + 1) * WP_ + (x + 1);
    const int hi = lane >> 4;
    const _Float16* bbase = in + p * CSTR + hi * 16;

    // stage tap 0 weights
    {
        const v8h* src = (const v8h*)wp;
        v8h* dst = (v8h*)lds[0];
        for (int i = tid; i < TAPV8; i += 256) dst[i] = src[i];
    }
    __syncthreads();

    v8f acc[NMT] = {};

    for (int tap = 0; tap < 9; ++tap) {
        // prefetch next tap's weight fragments into the other LDS buffer;
        // these global loads overlap the WMMA chain below.
        if (tap < 8) {
            const v8h* src = (const v8h*)(wp + (size_t)(tap + 1) * TAPH);
            v8h* dst = (v8h*)lds[(tap + 1) & 1];
            for (int i = tid; i < TAPV8; i += 256) dst[i] = src[i];
        }
        const int dy = tap / 3 - 1;
        const int dx = tap - (tap / 3) * 3 - 1;
        const _Float16* bp = bbase + (long long)(dy * WP_ + dx) * CSTR;
        const _Float16* abuf = lds[tap & 1] + lane * 16;

        v16h bf[NCH];
#pragma unroll
        for (int c = 0; c < NCH; ++c) bf[c] = *(const v16h*)(bp + c * 32);

        // A-fragment stream from LDS, even/odd double-buffered; the
        // sched_group_barrier pattern [1 WMMA][2 DS-read] forces fragment
        // k+1's loads to issue right after WMMA k, overlapping its latency.
        v16h afA = *(const v16h*)(abuf);
        v16h afB;
#pragma unroll
        for (int k = 0; k < KFRAG; ++k) {
            if ((k & 1) == 0) {
                if (k + 1 < KFRAG) afB = *(const v16h*)(abuf + (k + 1) * 512);
                acc[k % NMT] = __builtin_amdgcn_wmma_f32_16x16x32_f16(
                    false, afA, false, bf[k / NMT], (short)0, acc[k % NMT],
                    false, false);
            } else {
                if (k + 1 < KFRAG) afA = *(const v16h*)(abuf + (k + 1) * 512);
                acc[k % NMT] = __builtin_amdgcn_wmma_f32_16x16x32_f16(
                    false, afB, false, bf[k / NMT], (short)0, acc[k % NMT],
                    false, false);
            }
            SGB(0x008, 1);   // one WMMA
            SGB(0x100, 2);   // then the two ds_load_b128 of the next fragment
        }
        __syncthreads();
    }

    const int cb = hi * 8;
    _Float16* op = out + p * CSTR;
#pragma unroll
    for (int m = 0; m < NMT; ++m) {
        v8h h;
#pragma unroll
        for (int r = 0; r < 8; ++r) {
            int co = m * 16 + cb + r;
            float v = acc[m][r] + (co < Cout ? bias[co] : 0.f);
            v = v >= 0.f ? v : 0.1f * v;
            h[r] = (_Float16)v;
        }
        *(v8h*)(op + m * 16 + cb) = h;
    }
}

// ---------------------------------------------------------------------------
// Dynamic filter: softmax(-x^2) over 9 channels, applied to replication-
// padded occ patches (dy-major, dx-minor — matches Unfold order).
// ---------------------------------------------------------------------------
__global__ __launch_bounds__(256)
void dyn_filter_kernel(const _Float16* __restrict__ feat,
                       const float* __restrict__ occ,
                       float* __restrict__ out)
{
    int q  = blockIdx.x * 256 + threadIdx.x;
    int x  = q & (W_ - 1);
    int yb = q >> 9;
    int y  = yb & (H_ - 1);
    int b  = yb >> 8;
    size_t p = (size_t)(b * HP_ + y + 1) * WP_ + (x + 1);
    const _Float16* f = feat + p * CSTR;
    float wgt[9], s = 0.f;
#pragma unroll
    for (int k = 0; k < 9; ++k) {
        float v = (float)f[k];
        float e = __expf(-v * v);
        wgt[k] = e; s += e;
    }
    const float* ob = occ + (size_t)b * (H_ * W_);
    float r = 0.f;
#pragma unroll
    for (int dy = 0; dy < 3; ++dy) {
        int yy = min(max(y + dy - 1, 0), H_ - 1);
#pragma unroll
        for (int dx = 0; dx < 3; ++dx) {
            int xc = min(max(x + dx - 1, 0), W_ - 1);
            r += wgt[dy * 3 + dx] * ob[(size_t)yy * W_ + xc];
        }
    }
    out[q] = r / s;
}

// ---------------------------------------------------------------------------
static void run_conv(const _Float16* in, _Float16* out, const _Float16* wp,
                     const float* bias, int Cout, int nch, int nmt,
                     hipStream_t s)
{
#define CONV_CASE(C, M)                                                        \
    if (nch == C && nmt == M) {                                                \
        conv_kernel<C, M><<<NPIX / 128, 256, 0, s>>>(in, out, wp, bias, Cout); \
        return;                                                                \
    }
    CONV_CASE(3, 8)
    CONV_CASE(4, 8)
    CONV_CASE(4, 4)
    CONV_CASE(2, 4)
    CONV_CASE(2, 2)
    CONV_CASE(1, 2)
    CONV_CASE(1, 1)
#undef CONV_CASE
}

extern "C" void kernel_launch(void* const* d_in, const int* in_sizes, int n_in,
                              void* d_out, int out_size, void* d_ws, size_t ws_size,
                              hipStream_t stream)
{
    const float* occ = (const float*)d_in[0];
    const float* f1  = (const float*)d_in[1];
    const float* f2  = (const float*)d_in[2];

    char* ws = (char*)d_ws;
    const size_t BUF_BYTES = (size_t)PP_ * CSTR * sizeof(_Float16);
    _Float16* bufA = (_Float16*)ws;
    _Float16* bufB = (_Float16*)(ws + BUF_BYTES);
    _Float16* wpk  = (_Float16*)(ws + 2 * BUF_BYTES);

    static const int Cin[7]  = {65, 128, 128, 64, 64, 32, 32};
    static const int Cout[7] = {128, 128, 64, 64, 32, 32, 9};
    static const int NCH[7]  = {3, 4, 4, 2, 2, 1, 1};   /* Cin_pad / 32  */
    static const int NMT[7]  = {8, 8, 4, 4, 2, 2, 1};   /* Cout_pad / 16 */

    size_t woff[7]; size_t o = 0;
    for (int l = 0; l < 7; ++l) { woff[l] = o; o += (size_t)9 * NCH[l] * NMT[l] * 512; }

    init_pack_kernel<<<(PP_ + 255) / 256, 256, 0, stream>>>(occ, f1, f2, bufA, bufB);

    for (int l = 0; l < 7; ++l) {
        int total = 9 * NCH[l] * NMT[l] * 512;
        pack_w_kernel<<<(total + 255) / 256, 256, 0, stream>>>(
            (const float*)d_in[3 + 2 * l], wpk + woff[l],
            Cin[l], Cout[l], NCH[l], NMT[l], total);
    }

    _Float16* bin  = bufA;
    _Float16* bout = bufB;
    for (int l = 0; l < 7; ++l) {
        run_conv(bin, bout, wpk + woff[l], (const float*)d_in[4 + 2 * l],
                 Cout[l], NCH[l], NMT[l], stream);
        _Float16* t = bin; bin = bout; bout = t;
    }

    dyn_filter_kernel<<<NPIX / 256, 256, 0, stream>>>(bin, occ, (float*)d_out);
}
